// MultiheadAttetion_28853590295111
// MI455X (gfx1250) — matrix operand order repaired and checked by reading, not compiled, not run on hardware
//
#include <hip/hip_runtime.h>
#include <math.h>

#define DEV_INLINE __device__ __forceinline__

constexpr int B_ = 2, S_ = 2048, D_ = 1024, H_ = 16, HD_ = 64;
constexpr int BM = 128, BN = 64, BK = 32;  // block tile 128x64, K-step 32; 8 waves of 32x32
constexpr int LSTR = 40;                   // LDS row stride in ushorts: 80B rows -> b128-aligned

typedef __attribute__((ext_vector_type(16))) __bf16 bf16x16;
typedef __attribute__((ext_vector_type(8)))  float  f32x8;

union FragU { unsigned int u[8]; bf16x16 v; };

// Native bf16 convert (hardware v_cvt, RNE) instead of integer emulation.
DEV_INLINE unsigned short f32_to_bf16(float x) {
    union { __bf16 h; unsigned short u; } c;
    c.h = (__bf16)x;
    return c.u;
}

// ---- CDNA5 async global->LDS copy (ASYNCcnt path), 16 bytes per lane ----
DEV_INLINE void async_b128(unsigned lds_off, const void* g) {
    asm volatile("global_load_async_to_lds_b128 %0, %1, off"
                 :: "v"(lds_off), "v"((unsigned long long)(uintptr_t)g)
                 : "memory");
}
DEV_INLINE void wait_async0() { asm volatile("s_wait_asynccnt 0" ::: "memory"); }

// A fragment: 16x32 bf16 (ISA 7.12.2). Lane half selects K-octet pairing.
DEV_INLINE bf16x16 frag_a(const unsigned short* lds, int rowOff, int lane) {
    FragU f;
    const unsigned short* row = lds + (rowOff + (lane & 15)) * LSTR;
    const int kb = (lane >> 4) << 3;  // 0 or 8
#pragma unroll
    for (int j = 0; j < 4; ++j) f.u[j]     = *(const unsigned int*)(row + kb + 2 * j);
#pragma unroll
    for (int j = 0; j < 4; ++j) f.u[4 + j] = *(const unsigned int*)(row + kb + 16 + 2 * j);
    return f.v;
}

// B fragment: 32x16 bf16, lane half selects K-octet pairing (same as A); LDS tile stored [n][k].
DEV_INLINE bf16x16 frag_b(const unsigned short* lds, int colOff, int lane) {
    FragU f;
    const unsigned short* row = lds + (colOff + (lane & 15)) * LSTR;
    const int kb = (lane >> 4) << 3;  // 0 or 8
#pragma unroll
    for (int j = 0; j < 4; ++j) f.u[j]     = *(const unsigned int*)(row + kb + 2 * j);
#pragma unroll
    for (int j = 0; j < 4; ++j) f.u[4 + j] = *(const unsigned int*)(row + kb + 16 + 2 * j);
    return f.v;
}

DEV_INLINE f32x8 wmma_bf16(bf16x16 a, bf16x16 b, f32x8 c) {
    return __builtin_amdgcn_wmma_f32_16x16x32_bf16(false, a, false, b, (short)0, c, false, false);
}

// One K-step of a 32x32 wave tile: 2 A-frags x 2 B-frags -> 4 WMMAs.
DEV_INLINE void mma_step(const unsigned short* lA, const unsigned short* lB,
                         int rowOff, int colOff, int lane, f32x8 acc[4]) {
    bf16x16 a0 = frag_a(lA, rowOff,      lane);
    bf16x16 a1 = frag_a(lA, rowOff + 16, lane);
    bf16x16 b0 = frag_b(lB, colOff,      lane);
    bf16x16 b1 = frag_b(lB, colOff + 16, lane);
    acc[0] = wmma_bf16(a0, b0, acc[0]);
    acc[1] = wmma_bf16(a0, b1, acc[1]);
    acc[2] = wmma_bf16(a1, b0, acc[2]);
    acc[3] = wmma_bf16(a1, b1, acc[3]);
}

#define ACC4_INIT {{0.f,0.f,0.f,0.f,0.f,0.f,0.f,0.f},{0.f,0.f,0.f,0.f,0.f,0.f,0.f,0.f},\
                   {0.f,0.f,0.f,0.f,0.f,0.f,0.f,0.f},{0.f,0.f,0.f,0.f,0.f,0.f,0.f,0.f}}

// ---------------------------------------------------------------------------
// Kernel 1: P = X @ Wq^T  (NT GEMM, M=4096, N=1024, K=1024)
// Double-buffered LDS, one barrier per K-step; f32 -> bf16 via native cvt.
// ---------------------------------------------------------------------------
__global__ __launch_bounds__(256) void proj_kernel(const float* __restrict__ X,
                                                   const float* __restrict__ W,
                                                   unsigned short* __restrict__ Out) {
    __shared__ __align__(16) unsigned short lA[2][BM * LSTR];
    __shared__ __align__(16) unsigned short lB[2][BN * LSTR];
    const int tid = threadIdx.x, lane = tid & 31, wid = tid >> 5;
    const int waveM = wid >> 1, waveN = wid & 1;       // 4 x 2 waves, 32x32 each
    const int gm = blockIdx.y * BM, gn = blockIdx.x * BN;

    f32x8 acc[4] = ACC4_INIT;
    const int ra = tid >> 1, ca = (tid & 1) * 16;      // A: 128 x 32
    const int rb = tid >> 2, cb = (tid & 3) * 8;       // B: 64 x 32

    auto load_tile = [&](int k0, unsigned short* dA, unsigned short* dB) {
        const float* sa = X + (size_t)(gm + ra) * D_ + k0 + ca;
        unsigned short* da = dA + ra * LSTR + ca;
#pragma unroll
        for (int j = 0; j < 16; ++j) da[j] = f32_to_bf16(sa[j]);
        const float* sb = W + (size_t)(gn + rb) * D_ + k0 + cb;
        unsigned short* db = dB + rb * LSTR + cb;
#pragma unroll
        for (int j = 0; j < 8; ++j) db[j] = f32_to_bf16(sb[j]);
        __builtin_prefetch(sa + BK, 0, 1);             // global_prefetch_b8
    };

    load_tile(0, lA[0], lB[0]);
    int buf = 0;
    for (int k0 = 0; k0 < D_; k0 += BK) {
        __syncthreads();                               // tile[buf] visible; prior reads done
        if (k0 + BK < D_) load_tile(k0 + BK, lA[buf ^ 1], lB[buf ^ 1]);
        mma_step(lA[buf], lB[buf], waveM * 32, waveN * 32, lane, acc);
        buf ^= 1;
    }

    const int n = lane & 15, mb = (lane >> 4) << 3;
#pragma unroll
    for (int tm = 0; tm < 2; ++tm)
#pragma unroll
        for (int tn = 0; tn < 2; ++tn) {
            const int e = gn + waveN * 32 + tn * 16 + n;
            const int h = e >> 6, hd = e & (HD_ - 1);
#pragma unroll
            for (int j = 0; j < 8; ++j) {
                const int m = gm + waveM * 32 + tm * 16 + mb + j;
                const int bb = m >> 11, s = m & (S_ - 1);
                Out[(((size_t)bb * H_ + h) * S_ + s) * HD_ + hd] = f32_to_bf16(acc[tm * 2 + tn][j]);
            }
        }
}

// ---------------------------------------------------------------------------
// Kernel 2: logits = Q @ K^T / 8 + mask  (per b,h; M=N=S, K=HD) -> f32 scores
// bf16 tiles streamed with GLOBAL_LOAD_ASYNC_TO_LDS_B128, double-buffered.
// ---------------------------------------------------------------------------
__global__ __launch_bounds__(256) void logits_kernel(const unsigned short* __restrict__ Qh,
                                                     const unsigned short* __restrict__ Kh,
                                                     const float* __restrict__ mask,
                                                     float* __restrict__ scores) {
    __shared__ __align__(16) unsigned short lA[2][BM * LSTR];
    __shared__ __align__(16) unsigned short lB[2][BN * LSTR];
    const int tid = threadIdx.x, lane = tid & 31, wid = tid >> 5;
    const int waveM = wid >> 1, waveN = wid & 1;
    const int bh = blockIdx.z, b = bh >> 4;
    const int gm = blockIdx.y * BM, gn = blockIdx.x * BN;
    const unsigned short* Q = Qh + (size_t)bh * S_ * HD_;
    const unsigned short* K = Kh + (size_t)bh * S_ * HD_;

    f32x8 acc[4] = ACC4_INIT;
    const int ra = tid >> 1, ca = (tid & 1) * 16;      // A: 128 rows, 16 ushorts/thread
    const int rb = tid >> 2, cb = (tid & 3) * 8;       // B: 64 rows, 8 ushorts/thread

    auto load_tile = [&](int k0, unsigned short* dA, unsigned short* dB) {
        const unsigned short* sa = Q + (size_t)(gm + ra) * HD_ + k0 + ca;
        const unsigned lao = (unsigned)(uintptr_t)(dA + ra * LSTR + ca);
        async_b128(lao,      sa);
        async_b128(lao + 16, sa + 8);
        const unsigned short* sb = K + (size_t)(gn + rb) * HD_ + k0 + cb;
        async_b128((unsigned)(uintptr_t)(dB + rb * LSTR + cb), sb);
    };

    load_tile(0, lA[0], lB[0]);
    int buf = 0;
    for (int k0 = 0; k0 < HD_; k0 += BK) {
        wait_async0();                                 // this thread's copies for tile[buf] done
        __syncthreads();                               // all threads' copies visible
        if (k0 + BK < HD_) load_tile(k0 + BK, lA[buf ^ 1], lB[buf ^ 1]);
        mma_step(lA[buf], lB[buf], waveM * 32, waveN * 32, lane, acc);
        buf ^= 1;
    }

    const int n = lane & 15, mb = (lane >> 4) << 3;
#pragma unroll
    for (int tm = 0; tm < 2; ++tm)
#pragma unroll
        for (int tn = 0; tn < 2; ++tn) {
            const int kc = gn + waveN * 32 + tn * 16 + n;
#pragma unroll
            for (int j = 0; j < 8; ++j) {
                const int q = gm + waveM * 32 + tm * 16 + mb + j;
                float v = acc[tm * 2 + tn][j] * 0.125f;           // 1/sqrt(64)
                if (!(mask[((size_t)b * S_ + q) * S_ + kc] > 1e-6f)) v -= 1e6f;
                scores[((size_t)bh * S_ + q) * S_ + kc] = v;
            }
        }
}

// ---------------------------------------------------------------------------
// Kernel 3: in-place row softmax; one block per row of 2048
// ---------------------------------------------------------------------------
__global__ __launch_bounds__(256) void softmax_kernel(float* __restrict__ scores) {
    __shared__ float red[256];
    const int tid = threadIdx.x;
    float* p = scores + (size_t)blockIdx.x * S_;
    float v[8];
    float mx = -3.0e38f;
#pragma unroll
    for (int j = 0; j < 8; ++j) { v[j] = p[tid + j * 256]; mx = fmaxf(mx, v[j]); }
    red[tid] = mx; __syncthreads();
    for (int s = 128; s > 0; s >>= 1) { if (tid < s) red[tid] = fmaxf(red[tid], red[tid + s]); __syncthreads(); }
    mx = red[0]; __syncthreads();
    float sum = 0.f;
#pragma unroll
    for (int j = 0; j < 8; ++j) { v[j] = __expf(v[j] - mx); sum += v[j]; }
    red[tid] = sum; __syncthreads();
    for (int s = 128; s > 0; s >>= 1) { if (tid < s) red[tid] += red[tid + s]; __syncthreads(); }
    const float inv = 1.0f / red[0];
#pragma unroll
    for (int j = 0; j < 8; ++j) p[tid + j * 256] = v[j] * inv;
}

// ---------------------------------------------------------------------------
// Kernel 4: attn = scores @ V  (NN GEMM per b,h; M=S, N=HD=64, K=S)
// ---------------------------------------------------------------------------
__global__ __launch_bounds__(256) void attnv_kernel(const float* __restrict__ scores,
                                                    const unsigned short* __restrict__ Vh,
                                                    unsigned short* __restrict__ attnB) {
    __shared__ __align__(16) unsigned short lA[2][BM * LSTR];
    __shared__ __align__(16) unsigned short lB[2][BN * LSTR];
    const int tid = threadIdx.x, lane = tid & 31, wid = tid >> 5;
    const int waveM = wid >> 1, waveN = wid & 1;
    const int bh = blockIdx.z, b = bh >> 4, h = bh & 15;
    const int gm = blockIdx.y * BM, gn = blockIdx.x * BN;   // gn == 0 (BN == HD)
    const float* Sc = scores + (size_t)bh * S_ * S_;
    const unsigned short* V = Vh + (size_t)bh * S_ * HD_;

    f32x8 acc[4] = ACC4_INIT;
    const int ra = tid >> 1, ca = (tid & 1) * 16;       // A: 128 x 32 f32 -> bf16
    const int nb = tid & 63, kb8 = (tid >> 6) * 8;      // B tile stored [n][k]

    auto load_tile = [&](int k0, unsigned short* dA, unsigned short* dB) {
        const float* sa = Sc + (size_t)(gm + ra) * S_ + k0 + ca;
        unsigned short* da = dA + ra * LSTR + ca;
#pragma unroll
        for (int j = 0; j < 16; ++j) da[j] = f32_to_bf16(sa[j]);
        unsigned short* db = dB + nb * LSTR + kb8;
#pragma unroll
        for (int j = 0; j < 8; ++j) db[j] = V[(size_t)(k0 + kb8 + j) * HD_ + gn + nb];
        __builtin_prefetch(sa + BK, 0, 1);
    };

    load_tile(0, lA[0], lB[0]);
    int buf = 0;
    for (int k0 = 0; k0 < S_; k0 += BK) {
        __syncthreads();
        if (k0 + BK < S_) load_tile(k0 + BK, lA[buf ^ 1], lB[buf ^ 1]);
        mma_step(lA[buf], lB[buf], waveM * 32, waveN * 32, lane, acc);
        buf ^= 1;
    }

    const int n = lane & 15, mb = (lane >> 4) << 3;
#pragma unroll
    for (int tm = 0; tm < 2; ++tm)
#pragma unroll
        for (int tn = 0; tn < 2; ++tn) {
            const int col = gn + waveN * 32 + tn * 16 + n;  // within HD
#pragma unroll
            for (int j = 0; j < 8; ++j) {
                const int q = gm + waveM * 32 + tm * 16 + mb + j;
                attnB[((size_t)b * S_ + q) * D_ + h * HD_ + col] = f32_to_bf16(acc[tm * 2 + tn][j]);
            }
        }
}

// ---------------------------------------------------------------------------
// Kernel 5: out = attn @ Wo^T + bo  (NT GEMM, M=4096, N=1024, K=1024)
// attn tile: async b128 copies; Wo: f32 -> bf16 native cvt. Double-buffered.
// ---------------------------------------------------------------------------
__global__ __launch_bounds__(256) void outproj_kernel(const unsigned short* __restrict__ A,
                                                      const float* __restrict__ Wo,
                                                      const float* __restrict__ bo,
                                                      float* __restrict__ Out) {
    __shared__ __align__(16) unsigned short lA[2][BM * LSTR];
    __shared__ __align__(16) unsigned short lB[2][BN * LSTR];
    const int tid = threadIdx.x, lane = tid & 31, wid = tid >> 5;
    const int waveM = wid >> 1, waveN = wid & 1;
    const int gm = blockIdx.y * BM, gn = blockIdx.x * BN;

    f32x8 acc[4] = ACC4_INIT;
    const int ra = tid >> 1, ca = (tid & 1) * 16;
    const int rb = tid >> 2, cb = (tid & 3) * 8;

    auto load_tile = [&](int k0, unsigned short* dA, unsigned short* dB) {
        const unsigned short* sa = A + (size_t)(gm + ra) * D_ + k0 + ca;
        const unsigned lao = (unsigned)(uintptr_t)(dA + ra * LSTR + ca);
        async_b128(lao,      sa);
        async_b128(lao + 16, sa + 8);
        const float* sb = Wo + (size_t)(gn + rb) * D_ + k0 + cb;
        unsigned short* db = dB + rb * LSTR + cb;
#pragma unroll
        for (int j = 0; j < 8; ++j) db[j] = f32_to_bf16(sb[j]);
        __builtin_prefetch(sb + BK, 0, 1);
    };

    load_tile(0, lA[0], lB[0]);
    int buf = 0;
    for (int k0 = 0; k0 < D_; k0 += BK) {
        wait_async0();
        __syncthreads();
        if (k0 + BK < D_) load_tile(k0 + BK, lA[buf ^ 1], lB[buf ^ 1]);
        mma_step(lA[buf], lB[buf], waveM * 32, waveN * 32, lane, acc);
        buf ^= 1;
    }

    const int n = lane & 15, mb = (lane >> 4) << 3;
#pragma unroll
    for (int tm = 0; tm < 2; ++tm)
#pragma unroll
        for (int tn = 0; tn < 2; ++tn) {
            const int e = gn + waveN * 32 + tn * 16 + n;
            const float bias = bo[e];
#pragma unroll
            for (int j = 0; j < 8; ++j) {
                const int m = gm + waveM * 32 + tm * 16 + mb + j;
                Out[(size_t)m * D_ + e] = acc[tm * 2 + tn][j] + bias;
            }
        }
}

// ---------------------------------------------------------------------------
extern "C" void kernel_launch(void* const* d_in, const int* in_sizes, int n_in,
                              void* d_out, int out_size, void* d_ws, size_t ws_size,
                              hipStream_t stream) {
    (void)in_sizes; (void)n_in; (void)out_size; (void)ws_size;
    const float* query = (const float*)d_in[0];
    const float* key   = (const float*)d_in[1];
    const float* value = (const float*)d_in[2];
    const float* mask  = (const float*)d_in[3];
    const float* Wq    = (const float*)d_in[4];
    const float* Wo    = (const float*)d_in[5];
    const float* bo    = (const float*)d_in[6];

    float* out    = (float*)d_out;
    float* scores = out + (size_t)B_ * S_ * D_;   // scores live directly in d_out

    unsigned short* Qh    = (unsigned short*)d_ws;
    const size_t headElems = (size_t)B_ * H_ * S_ * HD_;   // 4M
    unsigned short* Kh    = Qh + headElems;
    unsigned short* Vh    = Kh + headElems;
    unsigned short* attnB = Vh + headElems;

    const dim3 blk(256);

    const dim3 gProj(D_ / BN, (B_ * S_) / BM);            // 16 x 32
    proj_kernel<<<gProj, blk, 0, stream>>>(query, Wq, Qh);
    proj_kernel<<<gProj, blk, 0, stream>>>(key,   Wq, Kh);
    proj_kernel<<<gProj, blk, 0, stream>>>(value, Wq, Vh);

    const dim3 gLog(S_ / BN, S_ / BM, B_ * H_);           // 32 x 16 x 32
    logits_kernel<<<gLog, blk, 0, stream>>>(Qh, Kh, mask, scores);

    softmax_kernel<<<dim3(B_ * H_ * S_), blk, 0, stream>>>(scores);

    const dim3 gAv(HD_ / BN, S_ / BM, B_ * H_);           // 1 x 16 x 32
    attnv_kernel<<<gAv, blk, 0, stream>>>(scores, Vh, attnB);

    const dim3 gOut(D_ / BN, (B_ * S_) / BM);             // 16 x 32
    outproj_kernel<<<gOut, blk, 0, stream>>>(attnB, Wo, bo, out);
}
